// LSTMDecoder_34385508172377
// MI455X (gfx1250) — matrix operand order repaired and checked by reading, not compiled, not run on hardware
//
#include <hip/hip_runtime.h>
#include <hip/hip_bf16.h>

// LSTM decoder for MI455X (gfx1250, wave32, WMMA).
//   prep:  gather act -> bf16, convert W_hh / Wa -> bf16, base = x@Wx^T + biases (f32),
//          init running h/c state (h/c live in the h_n/c_n slots of d_out).
//   scan:  128 sequential step kernels. Each step: fused GEMM
//          g[b, 4096] = base + [act_l | h]_bf16 @ [Wa | W_hh]^T_bf16   (M=64,N=4096,K=1536)
//          via v_wmma_f32_16x16x32_bf16 (two fully-unrolled phases with immediate-offset
//          loads), then gate elementwise math. h ping-pongs through two bf16 buffers.

typedef __attribute__((ext_vector_type(16))) __bf16 v16bf;
typedef __attribute__((ext_vector_type(8)))  __bf16 v8bf;
typedef __attribute__((ext_vector_type(8)))  float   v8f;

#define H_DIM 1024
#define FIN   512
#define FACT  512
#define GDIM  4096   // 4*H
#define LSEQ  128
#define BATCH 64

union BF16x16 { v16bf v; v8bf h[2]; };

__device__ __forceinline__ float fast_sigmoid(float x) {
  return 1.0f / (1.0f + __expf(-x));
}
__device__ __forceinline__ float fast_tanh(float x) {
  float e = __expf(2.0f * x);
  return (e - 1.0f) / (e + 1.0f);
}

// -------------------- prep kernels --------------------

// W_hh (4096x1024 f32) -> bf16, same row-major layout.
__global__ void cvt_whh_kernel(const float* __restrict__ whh_f, __bf16* __restrict__ out) {
  int i = blockIdx.x * 256 + threadIdx.x;           // 4096*1024 elements
  out[i] = (__bf16)whh_f[i];
}

// Wa = W_ih[:, 512:1024] (4096x512) -> bf16 row-major.
__global__ void cvt_wa_kernel(const float* __restrict__ wih, __bf16* __restrict__ out) {
  int i = blockIdx.x * 256 + threadIdx.x;           // 4096*512 elements
  int row = i >> 9, col = i & 511;
  out[i] = (__bf16)wih[row * (FIN + FACT) + FIN + col];
}

// act[(l,b), k] = masked rule_embed + masked token_embed, stored bf16.
__global__ void act_kernel(const int* __restrict__ pa, const float* __restrict__ ruleE,
                           const float* __restrict__ tokE, __bf16* __restrict__ act) {
  int i = blockIdx.x * 256 + threadIdx.x;           // 8192*512 elements
  int row = i >> 9, col = i & 511;
  int r = pa[row * 3 + 0];
  int t = pa[row * 3 + 1];
  float v = 0.0f;
  if (r != -1) v += ruleE[(size_t)r * FACT + col];
  if (t != -1) v += tokE[(size_t)t * FACT + col];
  act[i] = (__bf16)v;
}

// Running state init: h/c (f32) into the d_out h_n/c_n slots, bf16 copy of h.
__global__ void init_state_kernel(const float* __restrict__ h0, const float* __restrict__ c0,
                                  float* __restrict__ h_state, float* __restrict__ c_state,
                                  __bf16* __restrict__ h_bf0) {
  int i = blockIdx.x * 256 + threadIdx.x;           // 64*1024 elements
  float h = h0[i];
  h_state[i] = h;
  c_state[i] = c0[i];
  h_bf0[i] = (__bf16)h;
}

// base[b,g] = sum_k x[b,k] * W_ih[g,k] (k<512) + b_ih[g] + b_hh[g]  (f32, one-time)
__global__ void base_kernel(const float* __restrict__ x, const float* __restrict__ wih,
                            const float* __restrict__ b_ih, const float* __restrict__ b_hh,
                            float* __restrict__ base) {
  int g = blockIdx.x * 256 + threadIdx.x;           // 0..4095
  int b = blockIdx.y;                               // 0..63
  const float4* xr = reinterpret_cast<const float4*>(x + (size_t)b * FIN);
  const float4* wr = reinterpret_cast<const float4*>(wih + (size_t)g * (FIN + FACT));
  float s = 0.0f;
#pragma unroll 4
  for (int k = 0; k < FIN / 4; ++k) {
    float4 xv = xr[k], wv = wr[k];
    s += xv.x * wv.x + xv.y * wv.y + xv.z * wv.z + xv.w * wv.w;
  }
  base[(size_t)b * GDIM + g] = s + b_ih[g] + b_hh[g];
}

// -------------------- LSTM step kernel --------------------
// grid = 32 workgroups (each owns a 32-wide hidden slice j0..j0+31), block = 256 (8 waves).
// Wave w: mb = w&3 (16-row batch block), gate pair = w>>2. 4 accumulators of 16x16.
// Two phases: ksteps over act_l/Wa (K=512) then over h/W_hh (K=1024); per-phase base
// pointers are hoisted so every fragment load is base + immediate offset.
__global__ __launch_bounds__(256)
void lstm_step_kernel(const __bf16* __restrict__ act_l,    // (64 x 512)  rows = batch
                      const __bf16* __restrict__ h_bf,     // (64 x 1024)
                      const __bf16* __restrict__ wa,       // (4096 x 512)
                      const __bf16* __restrict__ whh,      // (4096 x 1024)
                      const float*  __restrict__ base,     // (64 x 4096)
                      float* __restrict__ c_state,         // (64 x 1024) in-place
                      float* __restrict__ h_state,         // (64 x 1024)
                      float* __restrict__ hs_l,            // (64 x 1024)
                      __bf16* __restrict__ h_bf_out)       // (64 x 1024) next-step h
{
  __shared__ float gbuf[4][BATCH][32];                     // 32 KB gate pre-activations

  const int tid   = threadIdx.x;
  const int wave  = tid >> 5;
  const int lane  = tid & 31;
  const int mb    = wave & 3;         // batch row block (16 rows)
  const int gp    = wave >> 2;        // 0 -> gates {i,f}, 1 -> gates {g,o}
  const int j0    = blockIdx.x * 32;  // hidden slice base
  const int lhalf = lane >> 4;        // 0 | 1
  const int l15   = lane & 15;

  v8f acc[4];
#pragma unroll
  for (int t = 0; t < 4; ++t)
#pragma unroll
    for (int r = 0; r < 8; ++r) acc[t][r] = 0.0f;

  const int m   = mb * 16 + l15;      // batch row this lane supplies for A
  const int hiA = lhalf << 4;         // A-layout K sub-offset (0 or 16 elements)
  const int hiB = lhalf << 4;         // B-layout K sub-offset (0 or 16 elements)

  // Hoisted per-tile weight-row base pointers (row r = output gate column).
  const __bf16* bpa[4];
  const __bf16* bph[4];
#pragma unroll
  for (int gi = 0; gi < 2; ++gi)
#pragma unroll
    for (int nb = 0; nb < 2; ++nb) {
      const int gate = gp * 2 + gi;
      const int r    = gate * H_DIM + j0 + nb * 16 + l15;
      bpa[gi * 2 + nb] = wa  + (size_t)r * FACT  + hiB;
      bph[gi * 2 + nb] = whh + (size_t)r * H_DIM + hiB;
    }
  const __bf16* apa = act_l + (size_t)m * FACT  + hiA;
  const __bf16* aph = h_bf  + (size_t)m * H_DIM + hiA;

  // ---- phase 1: K = 0..511 over [act_l, Wa] (16 k-steps of 32) ----
#pragma unroll
  for (int ks = 0; ks < 16; ++ks) {
    BF16x16 afrag;
    afrag.h[0] = *reinterpret_cast<const v8bf*>(apa + ks * 32);
    afrag.h[1] = *reinterpret_cast<const v8bf*>(apa + ks * 32 + 8);
#pragma unroll
    for (int t = 0; t < 4; ++t) {
      const v16bf bfrag = *reinterpret_cast<const v16bf*>(bpa[t] + ks * 32);
      acc[t] = __builtin_amdgcn_wmma_f32_16x16x32_bf16(
          false, afrag.v, false, bfrag, (short)0, acc[t], false, false);
    }
  }

  // ---- phase 2: K = 512..1535 over [h, W_hh] (32 k-steps of 32) ----
#pragma unroll
  for (int ks = 0; ks < 32; ++ks) {
    BF16x16 afrag;
    afrag.h[0] = *reinterpret_cast<const v8bf*>(aph + ks * 32);
    afrag.h[1] = *reinterpret_cast<const v8bf*>(aph + ks * 32 + 8);
#pragma unroll
    for (int t = 0; t < 4; ++t) {
      const v16bf bfrag = *reinterpret_cast<const v16bf*>(bph[t] + ks * 32);
      acc[t] = __builtin_amdgcn_wmma_f32_16x16x32_bf16(
          false, afrag.v, false, bfrag, (short)0, acc[t], false, false);
    }
  }

  // C layout: vgpr r -> M = mb*16 + lhalf*8 + r, N = nb*16 + (lane&15)
#pragma unroll
  for (int gi = 0; gi < 2; ++gi) {
#pragma unroll
    for (int nb = 0; nb < 2; ++nb) {
      const int gate = gp * 2 + gi;
      const int n    = nb * 16 + l15;
#pragma unroll
      for (int r = 0; r < 8; ++r) {
        const int M = mb * 16 + lhalf * 8 + r;
        gbuf[gate][M][n] = acc[gi * 2 + nb][r];
      }
    }
  }
  __syncthreads();

  // elementwise gates over the 64x32 slice (2048 cells / 256 threads)
#pragma unroll
  for (int p = 0; p < 8; ++p) {
    const int e  = tid + p * 256;
    const int b  = e >> 5;
    const int jl = e & 31;
    const int j  = j0 + jl;
    const size_t bb = (size_t)b * GDIM;

    const float g_i = gbuf[0][b][jl] + base[bb + j];
    const float g_f = gbuf[1][b][jl] + base[bb + H_DIM + j];
    const float g_g = gbuf[2][b][jl] + base[bb + 2 * H_DIM + j];
    const float g_o = gbuf[3][b][jl] + base[bb + 3 * H_DIM + j];

    const float iv = fast_sigmoid(g_i);
    const float fv = fast_sigmoid(g_f);
    const float ov = fast_sigmoid(g_o);
    const float gv = fast_tanh(g_g);

    const int   hidx = b * H_DIM + j;
    const float c1 = fv * c_state[hidx] + iv * gv;
    const float h1 = ov * fast_tanh(c1);

    c_state[hidx]  = c1;
    h_state[hidx]  = h1;
    hs_l[hidx]     = h1;
    h_bf_out[hidx] = (__bf16)h1;
  }
}

// -------------------- launch --------------------

extern "C" void kernel_launch(void* const* d_in, const int* in_sizes, int n_in,
                              void* d_out, int out_size, void* d_ws, size_t ws_size,
                              hipStream_t stream) {
  const float* input_feature = (const float*)d_in[0];
  const int*   prev_actions  = (const int*)d_in[1];   // (L,B,3) int32
  // d_in[2] = mask (unused by the reference computation)
  const float* hidden_state  = (const float*)d_in[3];
  const float* state         = (const float*)d_in[4];
  const float* rule_embed    = (const float*)d_in[5];
  const float* token_embed   = (const float*)d_in[6];
  const float* W_ih          = (const float*)d_in[7];
  const float* W_hh          = (const float*)d_in[8];
  const float* b_ih          = (const float*)d_in[9];
  const float* b_hh          = (const float*)d_in[10];

  float* out     = (float*)d_out;
  float* hs      = out;                                    // (L,B,H)
  float* h_state = out + (size_t)LSEQ * BATCH * H_DIM;     // h_n — doubles as running h
  float* c_state = h_state + (size_t)BATCH * H_DIM;        // c_n — doubles as running c

  // workspace layout (bytes), total ~21.3 MB
  char* ws = (char*)d_ws;
  __bf16* act   = (__bf16*)(ws + 0);                       //  8 MB: (8192 x 512) bf16
  __bf16* whhb  = (__bf16*)(ws + 8388608);                 //  8 MB: (4096 x 1024) bf16
  __bf16* wab   = (__bf16*)(ws + 16777216);                //  4 MB: (4096 x 512) bf16
  float*  baseb = (float*) (ws + 20971520);                //  1 MB: (64 x 4096) f32
  __bf16* hbf0  = (__bf16*)(ws + 22020096);                // 128 KB ping
  __bf16* hbf1  = (__bf16*)(ws + 22151168);                // 128 KB pong

  cvt_whh_kernel<<<(GDIM * H_DIM) / 256, 256, 0, stream>>>(W_hh, whhb);
  cvt_wa_kernel <<<(GDIM * FACT) / 256, 256, 0, stream>>>(W_ih, wab);
  act_kernel    <<<(LSEQ * BATCH * FACT) / 256, 256, 0, stream>>>(prev_actions, rule_embed,
                                                                  token_embed, act);
  init_state_kernel<<<(BATCH * H_DIM) / 256, 256, 0, stream>>>(hidden_state, state,
                                                               h_state, c_state, hbf0);
  base_kernel<<<dim3(GDIM / 256, BATCH), 256, 0, stream>>>(input_feature, W_ih, b_ih, b_hh,
                                                           baseb);

  for (int l = 0; l < LSEQ; ++l) {
    const __bf16* hin  = (l & 1) ? hbf1 : hbf0;
    __bf16*       hout = (l & 1) ? hbf0 : hbf1;
    lstm_step_kernel<<<32, 256, 0, stream>>>(
        act + (size_t)l * BATCH * FACT, hin, wab, whhb, baseb,
        c_state, h_state, hs + (size_t)l * BATCH * H_DIM, hout);
  }
}
